// Indexer_59227599012425
// MI455X (gfx1250) — matrix operand order repaired and checked by reading, not compiled, not run on hardware
//
#include <hip/hip_runtime.h>
#include <hip/hip_bf16.h>
#include <cstdint>

// ---------------------------------------------------------------------------
// Types for CDNA5 WMMA
// ---------------------------------------------------------------------------
typedef __bf16 bf16_t;
typedef __attribute__((ext_vector_type(16))) __bf16 v16bf;
typedef __attribute__((ext_vector_type(8)))  __bf16 v8bf;
typedef __attribute__((ext_vector_type(8)))  float  v8f;
typedef __attribute__((ext_vector_type(4)))  int    v4i;

#define SEQ   2048
#define DIM   7168
#define QLR   1536
#define NH    64
#define HD    128
#define NHD   8192   // NH*HD

// CDNA5 async global->LDS staging (ASYNCcnt-tracked), with portable fallback.
#if defined(__HIP_DEVICE_COMPILE__) && defined(__has_builtin)
#if __has_builtin(__builtin_amdgcn_global_load_async_to_lds_b128) && \
    __has_builtin(__builtin_amdgcn_s_wait_asynccnt)
#define USE_ASYNC_STAGE 1
#endif
#endif
#ifndef USE_ASYNC_STAGE
#define USE_ASYNC_STAGE 0
#endif

#define AS1 __attribute__((address_space(1)))
#define AS3 __attribute__((address_space(3)))

__device__ __forceinline__ void stage16(const unsigned short* g, unsigned short* l) {
#if USE_ASYNC_STAGE
  // builtin expects (int4 global*, int4 lds*, imm offset, imm cpol);
  // integer-cast idiom produces correctly address-spaced pointers.
  __builtin_amdgcn_global_load_async_to_lds_b128(
      (AS1 v4i*)(uintptr_t)g, (AS3 v4i*)(uintptr_t)l, 0, 0);
#else
  *(uint4*)l = *(const uint4*)g;   // ds_store_b128 fallback
#endif
}
__device__ __forceinline__ void stage_wait_keep1() {
#if USE_ASYNC_STAGE
  __builtin_amdgcn_s_wait_asynccnt(1);
#endif
}
__device__ __forceinline__ void stage_wait_all() {
#if USE_ASYNC_STAGE
  __builtin_amdgcn_s_wait_asynccnt(0);
#endif
}

__device__ __forceinline__ unsigned short f32_to_bf16(float f) {
  unsigned int u = __float_as_uint(f);
  u += 0x7FFFu + ((u >> 16) & 1u);   // round-to-nearest-even
  return (unsigned short)(u >> 16);
}
__device__ __forceinline__ float bf16_to_f32(unsigned short h) {
  return __uint_as_float(((unsigned int)h) << 16);
}

// A fragment: 16(M) x 32(K) bf16.  lane<16: M=lane, K in [0..8)∪[16..24);
// lane>=16: M=lane-16, K in [8..16)∪[24..32).  (ISA 7.12.2)
__device__ __forceinline__ v16bf load_a_frag(const bf16_t* base, int ld, int lane) {
  int m  = lane & 15;
  int hi = lane >> 4;
  const bf16_t* p = base + (size_t)m * ld + hi * 8;
  v8bf lo  = *(const v8bf*)(p);
  v8bf hi8 = *(const v8bf*)(p + 16);
  return __builtin_shufflevector(lo, hi8, 0,1,2,3,4,5,6,7,8,9,10,11,12,13,14,15);
}

// B fragment: 32(K) x 16(N) bf16, memory holds K contiguous per column N
// (base + n*ld + k).  lane<16: N=lane, K in [0..8)∪[16..24); lane>=16: N=lane-16, K in [8..16)∪[24..32).
__device__ __forceinline__ v16bf load_b_frag(const bf16_t* base, int ld, int lane) {
  int n    = lane & 15;
  int koff = (lane >> 4) * 8;
  const bf16_t* p = base + (size_t)n * ld + koff;
  v8bf lo  = *(const v8bf*)(p);
  v8bf hi8 = *(const v8bf*)(p + 16);
  return __builtin_shufflevector(lo, hi8, 0,1,2,3,4,5,6,7,8,9,10,11,12,13,14,15);
}

__device__ __forceinline__ v8f wmma_bf16(v16bf a, v16bf b, v8f c) {
  return __builtin_amdgcn_wmma_f32_16x16x32_bf16(false, a, false, b,
                                                 (short)0, c, false, false);
}

// ---------------------------------------------------------------------------
// Elementwise conversion kernels
// ---------------------------------------------------------------------------
__global__ void cvt_f32_bf16_kernel(const float* __restrict__ in,
                                    unsigned short* __restrict__ out, int n4) {
  int i = (blockIdx.x * blockDim.x + threadIdx.x);
  if (i < n4) {
    float4 f = ((const float4*)in)[i];
    int o = i * 4;
    out[o + 0] = f32_to_bf16(f.x);
    out[o + 1] = f32_to_bf16(f.y);
    out[o + 2] = f32_to_bf16(f.z);
    out[o + 3] = f32_to_bf16(f.w);
  }
}

__global__ void build_wcat_kernel(const float* __restrict__ wk,
                                  const float* __restrict__ wp,
                                  unsigned short* __restrict__ wcat) {
  int i = blockIdx.x * blockDim.x + threadIdx.x;   // over 192*7168
  if (i < 192 * DIM) {
    int row = i / DIM, col = i - row * DIM;
    float v = (row < HD) ? wk[(size_t)row * DIM + col]
                         : wp[(size_t)(row - HD) * DIM + col];
    wcat[i] = f32_to_bf16(v);
  }
}

// ---------------------------------------------------------------------------
// GEMM: KW[2048 x 192] = x_bf16 (2048x7168) @ wcat^T   (f32 out)
// ---------------------------------------------------------------------------
__global__ __launch_bounds__(128) void gemm_kw_kernel(
    const unsigned short* __restrict__ xb,
    const unsigned short* __restrict__ wcat,
    float* __restrict__ KW) {
  int lane = threadIdx.x & 31;
  int wave = threadIdx.x >> 5;
  int mtile = blockIdx.x;                 // 0..127
  int ntile = blockIdx.y * 4 + wave;      // 0..11
  const bf16_t* A0 = (const bf16_t*)xb   + (size_t)(mtile * 16) * DIM;
  const bf16_t* B0 = (const bf16_t*)wcat + (size_t)(ntile * 16) * DIM;
  v8f c = {};
  for (int k = 0; k < DIM; k += 32) {
    v16bf a = load_a_frag(A0 + k, DIM, lane);
    v16bf b = load_b_frag(B0 + k, DIM, lane);
    c = wmma_bf16(a, b, c);
  }
  int n  = lane & 15;
  int mb = (lane >> 4) * 8;
#pragma unroll
  for (int r = 0; r < 8; ++r)
    KW[(size_t)(mtile * 16 + mb + r) * 192 + ntile * 16 + n] = c[r];
}

// ---------------------------------------------------------------------------
// GEMM: Q[2048 x 8192] (bf16 out, pre-RoPE) = qr_bf16 @ wq_b_bf16^T
// 32x32 tile per wave (2x2 register blocking: 4 WMMAs per 2A+2B fragments)
// ---------------------------------------------------------------------------
__global__ __launch_bounds__(256) void gemm_q_kernel(
    const unsigned short* __restrict__ qrb,
    const unsigned short* __restrict__ wqb,
    unsigned short* __restrict__ Qb) {
  int lane = threadIdx.x & 31;
  int wave = threadIdx.x >> 5;
  int m0 = blockIdx.x * 32;                 // 64 m-macrotiles
  int n0 = (blockIdx.y * 8 + wave) * 32;    // 256 n-macrotiles
  const bf16_t* A0 = (const bf16_t*)qrb + (size_t)m0 * QLR;
  const bf16_t* A1 = A0 + (size_t)16 * QLR;
  const bf16_t* B0 = (const bf16_t*)wqb + (size_t)n0 * QLR;
  const bf16_t* B1 = B0 + (size_t)16 * QLR;
  v8f c00 = {}, c01 = {}, c10 = {}, c11 = {};
  for (int k = 0; k < QLR; k += 32) {
    v16bf a0 = load_a_frag(A0 + k, QLR, lane);
    v16bf a1 = load_a_frag(A1 + k, QLR, lane);
    v16bf b0 = load_b_frag(B0 + k, QLR, lane);
    v16bf b1 = load_b_frag(B1 + k, QLR, lane);
    c00 = wmma_bf16(a0, b0, c00);
    c01 = wmma_bf16(a0, b1, c01);
    c10 = wmma_bf16(a1, b0, c10);
    c11 = wmma_bf16(a1, b1, c11);
  }
  int n  = lane & 15;
  int mb = (lane >> 4) * 8;
#pragma unroll
  for (int r = 0; r < 8; ++r) {
    size_t rm0 = (size_t)(m0 + mb + r) * NHD;
    size_t rm1 = (size_t)(m0 + 16 + mb + r) * NHD;
    Qb[rm0 + n0 + n]      = f32_to_bf16(c00[r]);
    Qb[rm0 + n0 + 16 + n] = f32_to_bf16(c01[r]);
    Qb[rm1 + n0 + n]      = f32_to_bf16(c10[r]);
    Qb[rm1 + n0 + 16 + n] = f32_to_bf16(c11[r]);
  }
}

// ---------------------------------------------------------------------------
// k finalize: LayerNorm + RoPE + fast Walsh-Hadamard (1/sqrt(128)) -> Kb bf16
// also extracts scaled w row -> WmatT[h][s].  One 128-thread block per row.
// ---------------------------------------------------------------------------
__global__ __launch_bounds__(128) void k_finalize_kernel(
    const float* __restrict__ KW,
    const float* __restrict__ knw, const float* __restrict__ knb,
    const float* __restrict__ fcos, const float* __restrict__ fsin,
    unsigned short* __restrict__ Kb, float* __restrict__ WmatT) {
  int r = blockIdx.x;
  int d = threadIdx.x;
  __shared__ float v[128];
  __shared__ float tmp[128];

  float val = KW[(size_t)r * 192 + d];
  if (d < 64)   // WmatT is [NH][SEQ] so the score kernel can load it vectorized
    WmatT[(size_t)d * SEQ + r] =
        KW[(size_t)r * 192 + 128 + d] * 0.011048543456039806f;

  tmp[d] = val; __syncthreads();
  for (int s = 64; s > 0; s >>= 1) { if (d < s) tmp[d] += tmp[d + s]; __syncthreads(); }
  float mu = tmp[0] * (1.0f / 128.0f);
  __syncthreads();
  float dv = val - mu;
  tmp[d] = dv * dv; __syncthreads();
  for (int s = 64; s > 0; s >>= 1) { if (d < s) tmp[d] += tmp[d + s]; __syncthreads(); }
  float var = tmp[0] * (1.0f / 128.0f);
  __syncthreads();

  v[d] = dv * rsqrtf(var + 1e-6f) * knw[d] + knb[d];
  __syncthreads();
  float nv;
  if (d < 64) {
    int i = d >> 1;
    float c = fcos[(size_t)r * 32 + i], s = fsin[(size_t)r * 32 + i];
    float a = v[d & ~1], b = v[d | 1];
    nv = (d & 1) ? (a * s + b * c) : (a * c - b * s);
  } else nv = v[d];
  __syncthreads(); v[d] = nv; __syncthreads();

  for (int dist = 1; dist < 128; dist <<= 1) {
    float a = v[d], b = v[d ^ dist];
    float w = (d & dist) ? (b - a) : (a + b);
    __syncthreads(); v[d] = w; __syncthreads();
  }
  Kb[(size_t)r * 128 + d] = f32_to_bf16(v[d] * 0.08838834764831845f);
}

// ---------------------------------------------------------------------------
// q transform (in-place on bf16 Q): RoPE + FWHT per (s,h) row of 128
// ---------------------------------------------------------------------------
__global__ __launch_bounds__(128) void q_transform_kernel(
    unsigned short* __restrict__ Qb,
    const float* __restrict__ fcos, const float* __restrict__ fsin) {
  int s = blockIdx.x, h = blockIdx.y, d = threadIdx.x;
  unsigned short* base = Qb + (size_t)s * NHD + h * HD;
  __shared__ float v[128];
  v[d] = bf16_to_f32(base[d]);
  __syncthreads();
  float nv;
  if (d < 64) {
    int i = d >> 1;
    float c = fcos[(size_t)s * 32 + i], sn = fsin[(size_t)s * 32 + i];
    float a = v[d & ~1], b = v[d | 1];
    nv = (d & 1) ? (a * sn + b * c) : (a * c - b * sn);
  } else nv = v[d];
  __syncthreads(); v[d] = nv; __syncthreads();
  for (int dist = 1; dist < 128; dist <<= 1) {
    float a = v[d], b = v[d ^ dist];
    float w = (d & dist) ? (b - a) : (a + b);
    __syncthreads(); v[d] = w; __syncthreads();
  }
  base[d] = f32_to_bf16(v[d] * 0.08838834764831845f);
}

// ---------------------------------------------------------------------------
// Fused logits + relu*w head reduction + causal mask -> score f32 [2048x2048]
// Block: 8 waves share one s-tile; per-head 16x128 Q tile is staged to LDS
// once per block (async double-buffered), K fragments live in registers.
// ---------------------------------------------------------------------------
__global__ __launch_bounds__(256) void score_kernel(
    const unsigned short* __restrict__ Qb,
    const unsigned short* __restrict__ Kb,
    const float* __restrict__ WmatT,
    float* __restrict__ score) {
  __shared__ alignas(16) unsigned short qtile[2][16 * HD];   // 2 x 4KB
  int tid  = threadIdx.x;
  int lane = tid & 31;
  int wave = tid >> 5;
  int sb = blockIdx.x * 16;
  int tb = (blockIdx.y * 8 + wave) * 16;

  // hoisted K fragments (shared by all 64 heads)
  const bf16_t* Kbb = (const bf16_t*)Kb + (size_t)tb * HD;
  v16bf bfrag[4];
#pragma unroll
  for (int kk = 0; kk < 4; ++kk)
    bfrag[kk] = load_b_frag(Kbb + kk * 32, HD, lane);

  // staging map: thread -> one 16B chunk of the 16x128 bf16 tile
  int rrow = tid >> 4;            // 0..15
  int col  = (tid & 15) * 8;      // 0..120
  const unsigned short* gbase = Qb + (size_t)(sb + rrow) * NHD + col;
  unsigned short* l0 = &qtile[0][rrow * HD + col];
  unsigned short* l1 = &qtile[1][rrow * HD + col];

  stage16(gbase, l0);             // head 0 -> buffer 0

  int n  = lane & 15;
  int mb = (lane >> 4) * 8;
  v8f sc = {};
  for (int h = 0; h < NH; ++h) {
    if (h + 1 < NH) {
      stage16(gbase + (size_t)(h + 1) * HD, ((h + 1) & 1) ? l1 : l0);
      stage_wait_keep1();         // current buffer's loads complete
    } else {
      stage_wait_all();
    }
    __syncthreads();              // whole tile visible to all waves

    const bf16_t* Ab = (const bf16_t*)&qtile[h & 1][0];
    v8f c = {};
#pragma unroll
    for (int kk = 0; kk < 4; ++kk) {
      v16bf a = load_a_frag(Ab + kk * 32, HD, lane);
      c = wmma_bf16(a, bfrag[kk], c);
    }
    const float* wrow = WmatT + (size_t)h * SEQ + sb + mb;
    float4 w0 = *(const float4*)(wrow);
    float4 w1 = *(const float4*)(wrow + 4);
    sc[0] += fmaxf(c[0], 0.0f) * w0.x;
    sc[1] += fmaxf(c[1], 0.0f) * w0.y;
    sc[2] += fmaxf(c[2], 0.0f) * w0.z;
    sc[3] += fmaxf(c[3], 0.0f) * w0.w;
    sc[4] += fmaxf(c[4], 0.0f) * w1.x;
    sc[5] += fmaxf(c[5], 0.0f) * w1.y;
    sc[6] += fmaxf(c[6], 0.0f) * w1.z;
    sc[7] += fmaxf(c[7], 0.0f) * w1.w;
    __syncthreads();              // done reading before buffer is overwritten
  }
#pragma unroll
  for (int r = 0; r < 8; ++r) {
    int s = sb + mb + r, t = tb + n;
    score[(size_t)s * SEQ + t] = (t <= s) ? sc[r] : -1e30f;
  }
}

// ---------------------------------------------------------------------------
// Full per-row argsort (descending score, ties -> ascending index) = top_k(S)
// ---------------------------------------------------------------------------
__global__ __launch_bounds__(1024) void sort_kernel(
    const float* __restrict__ score, int* __restrict__ out) {
  __shared__ float key[SEQ];
  __shared__ int   idx[SEQ];
  int row = blockIdx.x;
  int tid = threadIdx.x;
  for (int i = tid; i < SEQ; i += 1024) { key[i] = score[(size_t)row * SEQ + i]; idx[i] = i; }
  __syncthreads();
  for (int k = 2; k <= SEQ; k <<= 1) {
    for (int j = k >> 1; j > 0; j >>= 1) {
      for (int base = 0; base < SEQ; base += 1024) {
        int i = base + tid;
        int ixj = i ^ j;
        if (ixj > i) {
          bool up = ((i & k) == 0);
          float ka = key[i], kb = key[ixj];
          int   ia = idx[i], ib = idx[ixj];
          bool a_first = (ka > kb) || (ka == kb && ia < ib);
          if (up ? !a_first : a_first) {
            key[i] = kb; key[ixj] = ka;
            idx[i] = ib; idx[ixj] = ia;
          }
        }
      }
      __syncthreads();
    }
  }
  for (int i = tid; i < SEQ; i += 1024) out[(size_t)row * SEQ + i] = idx[i];
}

// ---------------------------------------------------------------------------
// Launcher
// ---------------------------------------------------------------------------
extern "C" void kernel_launch(void* const* d_in, const int* in_sizes, int n_in,
                              void* d_out, int out_size, void* d_ws, size_t ws_size,
                              hipStream_t stream) {
  const float* x     = (const float*)d_in[0];   // 2048x7168
  const float* qr    = (const float*)d_in[1];   // 2048x1536
  const float* wq_b  = (const float*)d_in[2];   // 8192x1536
  const float* wk_w  = (const float*)d_in[3];   // 128x7168
  const float* knw   = (const float*)d_in[4];   // 128
  const float* knb   = (const float*)d_in[5];   // 128
  const float* wproj = (const float*)d_in[6];   // 64x7168
  const float* fcos  = (const float*)d_in[7];   // 2048x32
  const float* fsin  = (const float*)d_in[8];   // 2048x32
  // d_in[9] (mask) applied analytically; d_in[10] (start_pos) == 0
  int* out = (int*)d_out;
  char* ws = (char*)d_ws;

  // workspace layout (bytes)
  const size_t OFF_XB    = 0;                         // 2048*7168 bf16
  const size_t OFF_QRB   = OFF_XB  + (size_t)SEQ * DIM * 2;
  const size_t OFF_WQB   = OFF_QRB + (size_t)SEQ * QLR * 2;
  const size_t OFF_WCAT  = OFF_WQB + (size_t)NHD * QLR * 2;
  const size_t OFF_KW    = OFF_WCAT + (size_t)192 * DIM * 2;
  const size_t OFF_KB    = OFF_KW   + (size_t)SEQ * 192 * 4;
  const size_t OFF_WMAT  = OFF_KB   + (size_t)SEQ * HD * 2;
  const size_t OFF_QB    = OFF_WMAT + (size_t)SEQ * NH * 4;
  const size_t OFF_SCORE = OFF_QB   + (size_t)SEQ * NHD * 2;

  unsigned short* xb    = (unsigned short*)(ws + OFF_XB);
  unsigned short* qrb   = (unsigned short*)(ws + OFF_QRB);
  unsigned short* wqb   = (unsigned short*)(ws + OFF_WQB);
  unsigned short* wcat  = (unsigned short*)(ws + OFF_WCAT);
  float*          KW    = (float*)(ws + OFF_KW);
  unsigned short* Kb    = (unsigned short*)(ws + OFF_KB);
  float*          WmatT = (float*)(ws + OFF_WMAT);
  unsigned short* Qb    = (unsigned short*)(ws + OFF_QB);
  float*          score = (float*)(ws + OFF_SCORE);

  // 1) fp32 -> bf16 conversions
  {
    int n4 = SEQ * DIM / 4;
    cvt_f32_bf16_kernel<<<(n4 + 255) / 256, 256, 0, stream>>>(x, xb, n4);
  }
  {
    int n4 = SEQ * QLR / 4;
    cvt_f32_bf16_kernel<<<(n4 + 255) / 256, 256, 0, stream>>>(qr, qrb, n4);
  }
  {
    int n4 = NHD * QLR / 4;
    cvt_f32_bf16_kernel<<<(n4 + 255) / 256, 256, 0, stream>>>(wq_b, wqb, n4);
  }
  {
    int n = 192 * DIM;
    build_wcat_kernel<<<(n + 255) / 256, 256, 0, stream>>>(wk_w, wproj, wcat);
  }

  // 2) KW = x @ [wk; wproj]^T  (WMMA bf16)
  gemm_kw_kernel<<<dim3(SEQ / 16, 3), 128, 0, stream>>>(xb, wcat, KW);

  // 3) k: LayerNorm + RoPE + Hadamard -> Kb;  w -> WmatT
  k_finalize_kernel<<<SEQ, 128, 0, stream>>>(KW, knw, knb, fcos, fsin, Kb, WmatT);

  // 4) Q = qr @ wq_b^T  (WMMA bf16, 32x32 per wave)
  gemm_q_kernel<<<dim3(SEQ / 32, 32), 256, 0, stream>>>(qrb, wqb, Qb);

  // 5) q: RoPE + Hadamard in-place per (s,h)
  q_transform_kernel<<<dim3(SEQ, NH), 128, 0, stream>>>(Qb, fcos, fsin);

  // 6) fused logits (64 heads, WMMA) + relu*w + causal mask, LDS-staged Q
  score_kernel<<<dim3(SEQ / 16, 16), 256, 0, stream>>>(Qb, Kb, WmatT, score);

  // 7) top_k(2048) == full descending argsort per row
  sort_kernel<<<SEQ, 1024, 0, stream>>>(score, out);
}
